// PatchConsistency_15977278341524
// MI455X (gfx1250) — compile-verified, matrix operand order checked by reading, not run on hardware
//
#include <hip/hip_runtime.h>

typedef __attribute__((ext_vector_type(16))) _Float16 v16h;
typedef __attribute__((ext_vector_type(8)))  float    v8f;
typedef __attribute__((ext_vector_type(4)))  float    v4f;

#define BATCH   64
#define GH      32
#define GW      32
#define DIM     768
#define NPIX    (GH * GW)
#define THRESH2 1.0f
#define TASKS_PER_BATCH 126   // 64 horizontal groups + 62 vertical groups
#define TOTAL_WAVES (BATCH * TASKS_PER_BATCH)   // 8064

// Accumulate 8 squared diffs (fp32 -> f16) into A-fragment slots [BASE..BASE+7].
template <int BASE>
__device__ __forceinline__ void sq8(const float* __restrict__ pa,
                                    const float* __restrict__ pb,
                                    int off, v16h& af) {
    v4f a0 = *(const v4f*)(pa + off);
    v4f a1 = *(const v4f*)(pa + off + 4);
    v4f b0 = *(const v4f*)(pb + off);
    v4f b1 = *(const v4f*)(pb + off + 4);
    float d;
    d = a0.x - b0.x; af[BASE + 0] = (_Float16)(d * d);
    d = a0.y - b0.y; af[BASE + 1] = (_Float16)(d * d);
    d = a0.z - b0.z; af[BASE + 2] = (_Float16)(d * d);
    d = a0.w - b0.w; af[BASE + 3] = (_Float16)(d * d);
    d = a1.x - b1.x; af[BASE + 4] = (_Float16)(d * d);
    d = a1.y - b1.y; af[BASE + 5] = (_Float16)(d * d);
    d = a1.z - b1.z; af[BASE + 6] = (_Float16)(d * d);
    d = a1.w - b1.w; af[BASE + 7] = (_Float16)(d * d);
}

// Pass 1: one wave computes 16 edge norms^2 via v_wmma_f32_16x16x32_f16
// (A = diff^2 tile, B = all-ones => D[m][*] = sum_k diff2[m][k]), then
// atomically ORs threshold bits into exc[0..NPIX) (horizontal) /
// exc[NPIX..2*NPIX) (vertical).
__global__ __launch_bounds__(256) void pass1_norms(const float* __restrict__ x,
                                                   unsigned int* __restrict__ exc) {
    const int lane   = threadIdx.x & 31;
    const int waveId = blockIdx.x * 8 + (threadIdx.x >> 5);
    const int b = waveId / TASKS_PER_BATCH;
    const int t = waveId % TASKS_PER_BATCH;

    const int m    = lane & 15;   // edge index within group (row M of A)
    const int half = lane >> 4;   // lane half selects K sub-range

    int isV, h, w0;
    if (t < 64) {               // horizontal edges: (h, w0+m) -> (h, w0+m+1)
        isV = 0; h = t >> 1; w0 = (t & 1) ? 15 : 0;
    } else {                    // vertical edges: (h, w0+m) -> (h+1, w0+m)
        int tv = t - 64;
        isV = 1; h = tv >> 1; w0 = (tv & 1) << 4;
    }

    const long pixIdx = (long)b * NPIX + (long)h * GW + (w0 + m);
    const float* pa = x + pixIdx * DIM;
    const float* pb = pa + (isV ? (GW * DIM) : DIM);

    const int kbase = half * 8;   // A layout: lanes 0-15 -> K base 0, lanes 16-31 -> K base 8

    v16h ones;
#pragma unroll
    for (int i = 0; i < 16; ++i) ones[i] = (_Float16)1.0f;

    v8f c = {};
#pragma unroll 2
    for (int chunk = 0; chunk < DIM / 32; ++chunk) {
        const int d0 = chunk * 32 + kbase;      // K = kbase..kbase+7
        const int d1 = d0 + 16;                 // K = 16+kbase..16+kbase+7
        v16h af;
        sq8<0>(pa, pb, d0, af);
        sq8<8>(pa, pb, d1, af);
        c = __builtin_amdgcn_wmma_f32_16x16x32_f16(
                /*neg_a=*/false, af, /*neg_b=*/false, ones,
                /*c_mod=*/(short)0, c, /*reuse_a=*/false, /*reuse_b=*/false);
    }

    // C layout: VGPR j, lanes 0-15 -> row j; lanes 16-31 -> row j+8.
    if (m == 0) {
        unsigned int* arr = exc + isV * NPIX;
#pragma unroll
        for (int j = 0; j < 8; ++j) {
            if (c[j] > THRESH2) {
                int edge = j + 8 * half;
                atomicOr(&arr[h * GW + w0 + edge], 1u);
            }
        }
    }
}

// Pass 2: one 192-thread block per pixel; mask lookups are block-uniform
// (scalar loads); b128 streaming loads (RT: exploit L2 reuse of x) and
// non-temporal stores (output is write-once, keep x resident in the 192MB L2).
__global__ __launch_bounds__(192) void pass2_blend(const float* __restrict__ x,
                                                   float* __restrict__ out,
                                                   const unsigned int* __restrict__ exc) {
    const int pixel = blockIdx.x;          // b*NPIX + h*GW + w
    const int c4    = threadIdx.x;         // 0..191 float4 chunks of D
    const int w = pixel & (GW - 1);
    const int h = (pixel >> 5) & (GH - 1);

    const unsigned int* eh = exc;
    const unsigned int* ev = exc + NPIX;

    bool mask = false;
    if (h >= 1 && h <= GH - 2 && w >= 1 && w <= GW - 2) {
        unsigned int mm = eh[h * GW + (w - 1)] | eh[h * GW + w] |
                          ev[(h - 1) * GW + w] | ev[h * GW + w];
        mask = (mm != 0u);
    }

    const v4f* xb = (const v4f*)x + (long)pixel * (DIM / 4);
    v4f*       ob = (v4f*)out     + (long)pixel * (DIM / 4);

    v4f cv = xb[c4];
    v4f o;
    if (mask) {
        v4f u = xb[c4 - GW * (DIM / 4)];
        v4f d = xb[c4 + GW * (DIM / 4)];
        v4f l = xb[c4 - (DIM / 4)];
        v4f r = xb[c4 + (DIM / 4)];
        o = 0.5f * cv + 0.125f * (u + d + l + r);
    } else {
        o = cv;
    }
    // NT store: output is never re-read; don't evict x from L2.
    __builtin_nontemporal_store(o, &ob[c4]);
}

extern "C" void kernel_launch(void* const* d_in, const int* in_sizes, int n_in,
                              void* d_out, int out_size, void* d_ws, size_t ws_size,
                              hipStream_t stream) {
    const float* x   = (const float*)d_in[0];
    float*       out = (float*)d_out;
    unsigned int* exc = (unsigned int*)d_ws;   // 2 * NPIX uints = 8 KB

    (void)hipMemsetAsync(exc, 0, 2 * NPIX * sizeof(unsigned int), stream);

    pass1_norms<<<TOTAL_WAVES / 8, 256, 0, stream>>>(x, exc);
    pass2_blend<<<BATCH * NPIX, 192, 0, stream>>>(x, out, exc);
}
